// PosEvalModel_9131100472049
// MI455X (gfx1250) — matrix operand order repaired and checked by reading, not compiled, not run on hardware
//
#include <hip/hip_runtime.h>

// ---------------------------------------------------------------------------
// CDNA5 (gfx1250, wave32) WMMA types
// ---------------------------------------------------------------------------
typedef __attribute__((ext_vector_type(16))) __bf16 v16bf;
typedef __attribute__((ext_vector_type(8)))  float  v8f;

#define WAVE 32

// Problem sizes
#define B_   64
#define C_   50
#define L_   128
#define S_   128
#define G3S  384
#define N_   (B_ * C_)
#define V_   100008

// LDS pitches (elements)
#define XPITCH 136        // activation rows [16][136] bf16 (256B data + 16B pad)
#define G1P    392        // gate staging [16][392] f32
#define G2P    136        // h_n staging  [16][136] f32

__device__ __forceinline__ __bf16 to_bf16(float f) {
  unsigned int u = __builtin_bit_cast(unsigned int, f);
  unsigned int r = u + 0x7FFFu + ((u >> 16) & 1u);   // round-to-nearest-even
  unsigned short s = (unsigned short)(r >> 16);
  return __builtin_bit_cast(__bf16, s);
}

__device__ __forceinline__ float sigmoidf_(float x) {
  return 1.0f / (1.0f + __expf(-x));
}

// ---- CDNA5 async global->LDS copy (ASYNCcnt) --------------------------------
__device__ __forceinline__ void async_copy_b128(unsigned lds_byte_addr,
                                                const void* gaddr) {
  asm volatile("global_load_async_to_lds_b128 %0, %1, off"
               :: "v"(lds_byte_addr), "v"(gaddr) : "memory");
}
__device__ __forceinline__ void wait_async0() {
  asm volatile("s_wait_asynccnt 0x0" ::: "memory");
}

// A fragment: 16x32 bf16 tile from LDS, rows = M, cols = K from k0.
// ISA layout: lanes 0-15 -> M, K = (v/4)*16 + grp*8 + (v%4)*2 + hi
// -> two contiguous 16B chunks per lane; compiler lowers to 2x ds_load_b128.
__device__ __forceinline__ v16bf load_a_frag(const __bf16* base, int pitch,
                                             int lane, int k0) {
  const int grp = lane >> 4;
  const int m   = lane & 15;
  const __bf16* p = base + m * pitch + k0 + (grp << 3);
  v16bf a;
#pragma unroll
  for (int v = 0; v < 8; ++v) {
    const int k = ((v >> 2) << 4) + ((v & 3) << 1);
    a[2 * v]     = p[k];
    a[2 * v + 1] = p[k + 1];
  }
  return a;
}

// B fragment from pre-converted bf16 weights (row-major [ncols][pitch]).
// Element (k,n) = W[ncol0+n][k0+k]; lanes 0-15: K=0..7,16..23, lanes 16-31: K=8..15,24..31.
// -> two contiguous 16B runs per lane (pure loads, no conversion VALU).
__device__ __forceinline__ v16bf build_b_frag_bf16(const __bf16* w, int pitch,
                                                   int ncol0, int k0, int lane) {
  const int grp = lane >> 4;
  const int n   = lane & 15;
  const __bf16* p = w + (size_t)(ncol0 + n) * pitch + k0 + (grp << 3);
  v16bf b;
#pragma unroll
  for (int v = 0; v < 16; ++v) b[v] = p[((v >> 3) << 4) + (v & 7)];
  return b;
}

#define WMMA_BF16(a, b, c) \
  __builtin_amdgcn_wmma_f32_16x16x32_bf16(false, (a), false, (b), (short)0, (c), false, false)

// ---------------------------------------------------------------------------
// Streaming f32 -> bf16 conversion (embedding table + weight matrices).
// ---------------------------------------------------------------------------
__global__ void f32_to_bf16_kernel(const float* __restrict__ src,
                                   __bf16* __restrict__ dst, long long n) {
  long long i = ((long long)blockIdx.x * blockDim.x + threadIdx.x) * 8;
  const long long stride = (long long)gridDim.x * blockDim.x * 8;
  for (; i + 7 < n; i += stride) {
#pragma unroll
    for (int j = 0; j < 8; ++j) dst[i + j] = to_bf16(src[i + j]);
  }
}

// ---------------------------------------------------------------------------
// Persistent GRU over nsteps for a 16-row tile.
//   MODE 0 (AST): x[t] async-gathered from bf16 embed table via LDS token ids.
//   MODE 1 (CTX): x[t] async-copied from bf16 projs.
// B-operand WMMA fragments (both weight matrices, pre-converted bf16) live in
// VGPRs for the whole recurrence; h lives in LDS. x staging is double-buffered:
// step t+1's gather is issued (ASYNCcnt) before step t's WMMAs.
// 8 waves; wave w owns output column tiles {3w,3w+1,3w+2} of 24 (384/16).
// ---------------------------------------------------------------------------
template <int MODE>
__global__ void __launch_bounds__(256, 1)
gru_persistent_kernel(const int* __restrict__ token_ids,
                      const __bf16* __restrict__ embed_bf,
                      const __bf16* __restrict__ xsrc_bf,
                      const __bf16* __restrict__ w_ih_bf,
                      const __bf16* __restrict__ w_hh_bf,
                      const float* __restrict__ b_ih,
                      const float* __restrict__ b_hh,
                      const float* __restrict__ h_init,
                      int nsteps,
                      __bf16* __restrict__ hT_bf_out,
                      float* __restrict__ hT_f32_out) {
  __shared__ int    sTok[16 * L_];          // 8 KB   (MODE 0 only)
  __shared__ __bf16 sX[2][16 * XPITCH];     // 8.5 KB (double-buffered x)
  __shared__ __bf16 sHb[16 * XPITCH];       // 4.25 KB
  __shared__ float  sH[16 * S_];            // 8 KB
  __shared__ float  sG1[16 * G1P];          // 24.5 KB (r,z sums; i_n)
  __shared__ float  sG2[16 * G2P];          // 8.5 KB  (h_n)

  const int tid  = threadIdx.x;
  const int lane = tid & (WAVE - 1);
  const int wave = tid / WAVE;
  const int row0 = blockIdx.x * 16;

  // ---- register-resident weight fragments + biases (invariant over steps) ---
  v16bf bwih[3][4], bwhh[3][4];
  float bi[3], bh[3];
#pragma unroll
  for (int cc = 0; cc < 3; ++cc) {
    const int ncol0 = (wave * 3 + cc) * 16;
    bi[cc] = b_ih[ncol0 + (lane & 15)];
    bh[cc] = b_hh[ncol0 + (lane & 15)];
#pragma unroll
    for (int kk = 0; kk < 4; ++kk) {
      bwih[cc][kk] = build_b_frag_bf16(w_ih_bf, S_, ncol0, kk * 32, lane);
      bwhh[cc][kk] = build_b_frag_bf16(w_hh_bf, S_, ncol0, kk * 32, lane);
    }
  }

  // ---- token ids for the whole tile (contiguous block) + h init ------------
  if (MODE == 0) {
    for (int i = tid; i < 16 * L_; i += 256) sTok[i] = token_ids[row0 * L_ + i];
  }
  for (int i = tid; i < 16 * S_; i += 256) {
    const int m = i >> 7, k = i & 127;
    const float hv = h_init[k];
    sH[m * S_ + k]      = hv;
    sHb[m * XPITCH + k] = to_bf16(hv);
  }
  __syncthreads();

  // ---- async prefetch of x[0] into buffer 0 --------------------------------
  const int am    = tid >> 4;   // row within tile (0..15)
  const int chunk = tid & 15;   // 16B chunk within the 256B row (0..15)
  {
    const char* g;
    if (MODE == 0) {
      const int tok = sTok[am * L_ + 0];
      g = (const char*)(embed_bf + (size_t)tok * S_) + chunk * 16;
    } else {
      g = (const char*)(xsrc_bf + ((size_t)(row0 + am) * C_ + 0) * S_) + chunk * 16;
    }
    const unsigned loff =
        (unsigned)(unsigned long long)(&sX[0][0]) + am * (XPITCH * 2) + chunk * 16;
    async_copy_b128(loff, g);
  }
  wait_async0();
  __syncthreads();

  int buf = 0;
  for (int t = 0; t < nsteps; ++t) {
    // ---- issue async gather of x[t+1] into the other buffer ----------------
    if (t + 1 < nsteps) {
      const char* g;
      if (MODE == 0) {
        const int tok = sTok[am * L_ + (t + 1)];
        g = (const char*)(embed_bf + (size_t)tok * S_) + chunk * 16;
      } else {
        g = (const char*)(xsrc_bf + ((size_t)(row0 + am) * C_ + (t + 1)) * S_) +
            chunk * 16;
      }
      const unsigned loff =
          (unsigned)(unsigned long long)(&sX[buf ^ 1][0]) + am * (XPITCH * 2) +
          chunk * 16;
      async_copy_b128(loff, g);
    }

    // ---- gi = x @ Wih^T, gh = h @ Whh^T (bf16 WMMA, f32 accumulate) --------
    v8f ai[3], ah[3];
#pragma unroll
    for (int cc = 0; cc < 3; ++cc) {
      ai[cc] = (v8f){0.f, 0.f, 0.f, 0.f, 0.f, 0.f, 0.f, 0.f};
      ah[cc] = (v8f){0.f, 0.f, 0.f, 0.f, 0.f, 0.f, 0.f, 0.f};
    }
#pragma unroll
    for (int kk = 0; kk < 4; ++kk) {
      const int k0 = kk * 32;
      const v16bf ax = load_a_frag(&sX[buf][0], XPITCH, lane, k0);
      const v16bf av = load_a_frag(sHb, XPITCH, lane, k0);
#pragma unroll
      for (int cc = 0; cc < 3; ++cc) {
        ai[cc] = WMMA_BF16(ax, bwih[cc][kk], ai[cc]);
        ah[cc] = WMMA_BF16(av, bwhh[cc][kk], ah[cc]);
      }
    }

    // ---- stage gates: C/D layout VGPR r -> M = grp*8 + r, col = lane%16 ----
    const int n   = lane & 15;
    const int grp = lane >> 4;
#pragma unroll
    for (int cc = 0; cc < 3; ++cc) {
      const int c   = wave * 3 + cc;
      const int col = c * 16 + n;
#pragma unroll
      for (int r = 0; r < 8; ++r) {
        const int m = grp * 8 + r;
        const float vi = ai[cc][r] + bi[cc];
        const float vh = ah[cc][r] + bh[cc];
        if (c < 16) {                        // r / z gates: sum only
          sG1[m * G1P + col] = vi + vh;
        } else {                             // n gate: keep i_n and h_n split
          sG1[m * G1P + col]         = vi;
          sG2[m * G2P + (col - 256)] = vh;
        }
      }
    }
    wait_async0();       // x[t+1] landed in LDS (each wave waits its own ops)
    __syncthreads();     // staging + prefetch visible to everyone

    // ---- elementwise GRU update -------------------------------------------
    for (int i = tid; i < 16 * S_; i += 256) {
      const int m = i >> 7, j = i & 127;
      const float r  = sigmoidf_(sG1[m * G1P + j]);
      const float z  = sigmoidf_(sG1[m * G1P + j + 128]);
      const float nn = tanhf(sG1[m * G1P + j + 256] + r * sG2[m * G2P + j]);
      const float hv = (1.0f - z) * nn + z * sH[m * S_ + j];
      sH[m * S_ + j]      = hv;
      sHb[m * XPITCH + j] = to_bf16(hv);
    }
    __syncthreads();
    buf ^= 1;
  }

  // ---- write final hidden state --------------------------------------------
  for (int i = tid; i < 16 * S_; i += 256) {
    const int m = i >> 7, k = i & 127;
    if (MODE == 0) hT_bf_out[(row0 + m) * S_ + k] = sHb[m * XPITCH + k];
    else           hT_f32_out[(row0 + m) * S_ + k] = sH[m * S_ + k];
  }
}

// ---------------------------------------------------------------------------
// projs = [hT | indicator] @ proj_w^T + proj_b  (bf16 in, bf16 out)
// 128-K WMMA GEMM + rank-1 indicator correction (proj_w column 128 applies
// only to rows with row % C_ == 0).
// ---------------------------------------------------------------------------
__global__ void proj_kernel(const __bf16* __restrict__ hT,
                            const __bf16* __restrict__ proj_w_bf, // [128][129] bf16
                            const float* __restrict__ proj_w,     // [128][129] f32
                            const float* __restrict__ proj_b,     // [128]
                            __bf16* __restrict__ projs) {
  __shared__ __bf16 sX[16 * XPITCH];

  const int tid  = threadIdx.x;
  const int lane = tid & (WAVE - 1);
  const int wave = tid / WAVE;
  const int row0 = blockIdx.x * 16;

  for (int i = tid; i < 16 * S_; i += 256) {
    const int m = i >> 7, k = i & 127;
    sX[m * XPITCH + k] = hT[(row0 + m) * S_ + k];
  }

  const int ncol0 = wave * 16;          // 8 waves x 16 cols = 128
  v16bf bw[4];
#pragma unroll
  for (int kk = 0; kk < 4; ++kk)
    bw[kk] = build_b_frag_bf16(proj_w_bf, S_ + 1, ncol0, kk * 32, lane);
  __syncthreads();

  v8f acc = {0.f, 0.f, 0.f, 0.f, 0.f, 0.f, 0.f, 0.f};
#pragma unroll
  for (int kk = 0; kk < 4; ++kk)
    acc = WMMA_BF16(load_a_frag(sX, XPITCH, lane, kk * 32), bw[kk], acc);

  const int n   = lane & 15;
  const int grp = lane >> 4;
  const int col = ncol0 + n;
  const float bias = proj_b[col];
  const float wind = proj_w[col * (S_ + 1) + S_];   // indicator column (f32)
#pragma unroll
  for (int r = 0; r < 8; ++r) {
    const int row = row0 + grp * 8 + r;
    const float ind = ((row % C_) == 0) ? 1.0f : 0.0f;
    projs[row * S_ + col] = to_bf16(acc[r] + bias + ind * wind);
  }
}

// ---------------------------------------------------------------------------
// Final head: out[b,:] = cT[b,:] @ final_w^T + final_b   (64x3, trivial)
// ---------------------------------------------------------------------------
__global__ void final_head_kernel(const float* __restrict__ cT,
                                  const float* __restrict__ final_w,  // [3][128]
                                  const float* __restrict__ final_b,  // [3]
                                  float* __restrict__ out) {
  const int i = blockIdx.x * blockDim.x + threadIdx.x;  // 192 outputs
  if (i >= B_ * 3) return;
  const int b = i / 3, o = i % 3;
  float acc = final_b[o];
#pragma unroll 4
  for (int k = 0; k < S_; ++k) acc += cT[b * S_ + k] * final_w[o * S_ + k];
  out[b * 3 + o] = acc;
}

// ---------------------------------------------------------------------------
extern "C" void kernel_launch(void* const* d_in, const int* in_sizes, int n_in,
                              void* d_out, int out_size, void* d_ws, size_t ws_size,
                              hipStream_t stream) {
  const int*   token_ids = (const int*)d_in[0];
  const float* embed     = (const float*)d_in[1];
  const float* ast_w_ih  = (const float*)d_in[2];
  const float* ast_w_hh  = (const float*)d_in[3];
  const float* ast_b_ih  = (const float*)d_in[4];
  const float* ast_b_hh  = (const float*)d_in[5];
  const float* ast_init  = (const float*)d_in[6];
  const float* ctx_w_ih  = (const float*)d_in[7];
  const float* ctx_w_hh  = (const float*)d_in[8];
  const float* ctx_b_ih  = (const float*)d_in[9];
  const float* ctx_b_hh  = (const float*)d_in[10];
  const float* ctx_init  = (const float*)d_in[11];
  const float* proj_w    = (const float*)d_in[12];
  const float* proj_b    = (const float*)d_in[13];
  const float* final_w   = (const float*)d_in[14];
  const float* final_b   = (const float*)d_in[15];

  // workspace carve (all chunks 256B-aligned)
  char* ws = (char*)d_ws;
  size_t off = 0;
  __bf16* embed_bf   = (__bf16*)(ws + off); off += (size_t)V_ * S_ * 2;   // 25.6 MB
  __bf16* hT_bf      = (__bf16*)(ws + off); off += (size_t)N_ * S_ * 2;   // 800 KB
  __bf16* projs_bf   = (__bf16*)(ws + off); off += (size_t)N_ * S_ * 2;   // 800 KB
  float*  cT         = (float*) (ws + off); off += (size_t)B_ * S_ * 4;   // 32 KB
  __bf16* awih_bf    = (__bf16*)(ws + off); off += (size_t)G3S * S_ * 2;  // 96 KB
  __bf16* awhh_bf    = (__bf16*)(ws + off); off += (size_t)G3S * S_ * 2;
  __bf16* cwih_bf    = (__bf16*)(ws + off); off += (size_t)G3S * S_ * 2;
  __bf16* cwhh_bf    = (__bf16*)(ws + off); off += (size_t)G3S * S_ * 2;
  __bf16* projw_bf   = (__bf16*)(ws + off); off += (size_t)S_ * (S_ + 1) * 2;

  // 0) One-time f32 -> bf16 conversions (embedding table + weights).
  f32_to_bf16_kernel<<<4096, 256, 0, stream>>>(embed, embed_bf,
                                               (long long)V_ * S_);
  f32_to_bf16_kernel<<<24, 256, 0, stream>>>(ast_w_ih, awih_bf, (long long)G3S * S_);
  f32_to_bf16_kernel<<<24, 256, 0, stream>>>(ast_w_hh, awhh_bf, (long long)G3S * S_);
  f32_to_bf16_kernel<<<24, 256, 0, stream>>>(ctx_w_ih, cwih_bf, (long long)G3S * S_);
  f32_to_bf16_kernel<<<24, 256, 0, stream>>>(ctx_w_hh, cwhh_bf, (long long)G3S * S_);
  f32_to_bf16_kernel<<<9, 256, 0, stream>>>(proj_w, projw_bf,
                                            (long long)S_ * (S_ + 1));

  // 1) AST GRU: 200 persistent tiles of 16 sequences, 128 steps each.
  gru_persistent_kernel<0><<<N_ / 16, 256, 0, stream>>>(
      token_ids, embed_bf, nullptr,
      awih_bf, awhh_bf, ast_b_ih, ast_b_hh, ast_init, L_, hT_bf, nullptr);

  // 2) Projection with conclusion indicator.
  proj_kernel<<<N_ / 16, 256, 0, stream>>>(hT_bf, projw_bf, proj_w, proj_b,
                                           projs_bf);

  // 3) CTX GRU over the 50 context entries, 4 tiles of 16 batch rows.
  gru_persistent_kernel<1><<<B_ / 16, 256, 0, stream>>>(
      nullptr, nullptr, projs_bf,
      cwih_bf, cwhh_bf, ctx_b_ih, ctx_b_hh, ctx_init, C_, nullptr, cT);

  // 4) Final 128 -> 3 head.
  final_head_kernel<<<1, 256, 0, stream>>>(cT, final_w, final_b, (float*)d_out);
}
